// AdaptiveDirectionShift_13443247636654
// MI455X (gfx1250) — matrix operand order repaired and checked by reading, not compiled, not run on hardware
//
#include <hip/hip_runtime.h>
#include <hip/hip_bf16.h>

typedef __attribute__((ext_vector_type(16))) _Float16 v16h;
typedef __attribute__((ext_vector_type(8)))  float    v8f;
typedef __attribute__((ext_vector_type(4)))  float    v4f;

#define B_  16
#define C_  320
#define H_  64
#define W_  64
#define HW  (H_ * W_)          // 4096
#define NPIX (B_ * HW)         // 65536
#define NTOT (B_ * C_ * HW)    // 20971520

// ---------------------------------------------------------------------------
// Kernel 1: edge_map[b,p] = mean over 320 channels of edge_guidance.
// One thread per pixel (2048 waves); stride-16KB lane-coalesced reads,
// nontemporal (single-use stream, keep it out of L2), 4 accumulators for ILP.
// ---------------------------------------------------------------------------
__global__ void k_channel_mean(const float* __restrict__ eg,
                               float* __restrict__ em) {
    unsigned p   = blockIdx.x * blockDim.x + threadIdx.x;   // 0..65535
    unsigned b   = p >> 12;
    unsigned pix = p & (HW - 1);
    const float* base = eg + (size_t)b * C_ * HW + pix;
    float a0 = 0.f, a1 = 0.f, a2 = 0.f, a3 = 0.f;
    for (int c = 0; c < C_; c += 4) {
        a0 += __builtin_nontemporal_load(base + (size_t)(c + 0) * HW);
        a1 += __builtin_nontemporal_load(base + (size_t)(c + 1) * HW);
        a2 += __builtin_nontemporal_load(base + (size_t)(c + 2) * HW);
        a3 += __builtin_nontemporal_load(base + (size_t)(c + 3) * HW);
    }
    em[p] = (a0 + a1 + a2 + a3) * (1.0f / 320.0f);
}

// ---------------------------------------------------------------------------
// Kernel 2: conv1 (1 -> 16 ch, 3x3 SAME) + bias + relu via one WMMA per
// 16-pixel row strip.  D(16 ch x 16 pix) = A(16x32 weights) * B(32x16 patch).
// K = tap index 0..8, taps 9..31 zero-padded.
// ---------------------------------------------------------------------------
__global__ void k_conv1_wmma(const float* __restrict__ em,
                             const float* __restrict__ w1,
                             const float* __restrict__ b1,
                             _Float16* __restrict__ hws) {
    unsigned t    = blockIdx.x * blockDim.x + threadIdx.x;
    unsigned wave = t >> 5;            // 0..4095 strips: b(16) * y(64) * xblk(4)
    unsigned lane = t & 31;
    unsigned xblk = wave & 3;
    unsigned y    = (wave >> 2) & 63;
    unsigned b    = wave >> 8;
    int hi = lane >> 4;
    int mn = lane & 15;                // M for A rows, N (pixel) for B cols
    int x0 = (int)xblk * 16;

    v16h A, Bm;
    #pragma unroll
    for (int e = 0; e < 16; ++e) {
        // A-matrix (weights): lane row m = mn
        int kA = ((e >> 3) << 4) + hi * 8 + (e & 7);
        A[e] = (kA < 9) ? (_Float16)w1[mn * 9 + kA] : (_Float16)0.0f;
        // B-matrix (im2col patch): column n = mn (pixel x0+mn)
        int kB = ((e >> 3) << 4) + hi * 8 + (e & 7);
        _Float16 bv = (_Float16)0.0f;
        if (kB < 9) {
            int dy = kB / 3 - 1, dx = kB % 3 - 1;
            int yy = (int)y + dy, xx = x0 + mn + dx;
            bv = (yy >= 0 && yy < H_ && xx >= 0 && xx < W_)
                     ? (_Float16)em[b * HW + yy * W_ + xx]
                     : (_Float16)0.0f;
        }
        Bm[e] = bv;
    }
    v8f acc = {};
    acc = __builtin_amdgcn_wmma_f32_16x16x32_f16(
        false, A, false, Bm, (short)0, acc, false, false);

    #pragma unroll
    for (int r = 0; r < 8; ++r) {
        int ch = r + 8 * hi;                      // D: M = r + 8*hi, N = mn
        float v = acc[r] + b1[ch];
        v = v > 0.0f ? v : 0.0f;
        hws[((b * 16 + ch) * H_ + y) * W_ + x0 + mn] = (_Float16)v;
    }
}

// ---------------------------------------------------------------------------
// Kernel 3: conv2 (16 -> 2 ch, 3x3 SAME) + softmax over the 2 logits.
// K = ch*9 + tap, 0..143 padded to 160 -> 5 chained WMMAs per strip.
// Stores only w0 = softmax channel 0 (w1 = 1 - w0).
// ---------------------------------------------------------------------------
__global__ void k_conv2_softmax_wmma(const _Float16* __restrict__ hws,
                                     const float* __restrict__ w2,
                                     const float* __restrict__ b2,
                                     float* __restrict__ w0out) {
    unsigned t    = blockIdx.x * blockDim.x + threadIdx.x;
    unsigned wave = t >> 5;
    unsigned lane = t & 31;
    unsigned xblk = wave & 3;
    unsigned y    = (wave >> 2) & 63;
    unsigned b    = wave >> 8;
    int hi = lane >> 4;
    int mn = lane & 15;
    int x0 = (int)xblk * 16;

    v8f acc = {};
    #pragma unroll
    for (int cc = 0; cc < 5; ++cc) {
        v16h A, Bm;
        #pragma unroll
        for (int e = 0; e < 16; ++e) {
            int kA = cc * 32 + ((e >> 3) << 4) + hi * 8 + (e & 7);
            _Float16 av = (_Float16)0.0f;
            if (kA < 144 && mn < 2) {
                int ch = kA / 9, tap = kA % 9;
                av = (_Float16)w2[(mn * 16 + ch) * 9 + tap];
            }
            A[e] = av;
            int kB = cc * 32 + ((e >> 3) << 4) + hi * 8 + (e & 7);
            _Float16 bv = (_Float16)0.0f;
            if (kB < 144) {
                int ch = kB / 9, tap = kB % 9;
                int dy = tap / 3 - 1, dx = tap % 3 - 1;
                int yy = (int)y + dy, xx = x0 + mn + dx;
                if (yy >= 0 && yy < H_ && xx >= 0 && xx < W_)
                    bv = hws[((b * 16 + ch) * H_ + yy) * W_ + xx];
            }
            Bm[e] = bv;
        }
        acc = __builtin_amdgcn_wmma_f32_16x16x32_f16(
            false, A, false, Bm, (short)0, acc, false, false);
    }
    // logits live in M=0 (vgpr0) and M=1 (vgpr1) of lanes 0-15
    if (hi == 0) {
        float l0 = acc[0] + b2[0];
        float l1 = acc[1] + b2[1];
        float w0 = 1.0f / (1.0f + __expf(l1 - l0));   // softmax ch0
        w0out[b * HW + y * W_ + x0 + mn] = w0;
    }
}

// ---------------------------------------------------------------------------
// Kernel 4 (bandwidth stage): out = w0 * X[y-sh[g], x] + (1-w0) * X[y, x-sw[g]]
// 4 pixels per thread.  Aligned b128 for the row-shifted term / w0 / store
// (the row shift preserves x alignment); 4 scalar loads for the +-2-float
// column-shifted term (same cachelines, L0 hits).  Nontemporal store keeps
// the write stream from evicting x (84 MB, L2-resident and read twice).
// ---------------------------------------------------------------------------
__global__ void k_shift_combine(const float* __restrict__ x,
                                const float* __restrict__ w0,
                                const int* __restrict__ sh,
                                const int* __restrict__ sw,
                                float* __restrict__ out) {
    unsigned t  = blockIdx.x * blockDim.x + threadIdx.x;    // < NTOT/4
    int      xq = t & 15;               // quad index along W (16 quads/row)
    int      y  = (t >> 4) & 63;
    unsigned bc = t >> 10;
    int c = bc % C_;
    int b = bc / C_;
    int g = c >> 6;                     // 5 groups of 64 channels
    int s_h = sh[g];                    // uniform per block -> scalar loads
    int s_w = sw[g];
    int xw4 = xq << 2;

    const float* xb = x + ((size_t)bc << 12);

    // H-shifted term: whole quad shares the bounds check on ys
    int ys = y - s_h;
    v4f vh = {0.f, 0.f, 0.f, 0.f};
    if (ys >= 0 && ys < H_)
        vh = *(const v4f*)(xb + (ys << 6) + xw4);

    // W-shifted term: per-element column bounds
    const float* row = xb + (y << 6);
    v4f vw;
    #pragma unroll
    for (int j = 0; j < 4; ++j) {
        int xs = xw4 + j - s_w;
        vw[j] = (xs >= 0 && xs < W_) ? row[xs] : 0.0f;
    }

    v4f wv = *(const v4f*)(w0 + (b << 12) + (y << 6) + xw4);
    v4f res;
    #pragma unroll
    for (int j = 0; j < 4; ++j)
        res[j] = wv[j] * vh[j] + (1.0f - wv[j]) * vw[j];

    __builtin_nontemporal_store(res, (v4f*)(out + ((size_t)t << 2)));
}

// ---------------------------------------------------------------------------
extern "C" void kernel_launch(void* const* d_in, const int* in_sizes, int n_in,
                              void* d_out, int out_size, void* d_ws, size_t ws_size,
                              hipStream_t stream) {
    const float* x   = (const float*)d_in[0];
    const float* eg  = (const float*)d_in[1];
    const float* w1  = (const float*)d_in[2];
    const float* b1  = (const float*)d_in[3];
    const float* w2  = (const float*)d_in[4];
    const float* b2  = (const float*)d_in[5];
    const int*   shh = (const int*)d_in[6];
    const int*   shw = (const int*)d_in[7];
    float* out = (float*)d_out;

    // workspace: edge_map (256KB) | w0 (256KB) | h fp16 (2MB)
    float*     em  = (float*)d_ws;
    float*     w0  = em + NPIX;
    _Float16*  hws = (_Float16*)(w0 + NPIX);

    k_channel_mean<<<NPIX / 256, 256, 0, stream>>>(eg, em);
    k_conv1_wmma<<<(4096 * 32) / 256, 256, 0, stream>>>(em, w1, b1, hws);
    k_conv2_softmax_wmma<<<(4096 * 32) / 256, 256, 0, stream>>>(hws, w2, b2, w0);
    k_shift_combine<<<(NTOT / 4) / 256, 256, 0, stream>>>(x, w0, shh, shw, out);
}
